// AdaptiveAttention_85693187490437
// MI455X (gfx1250) — matrix operand order repaired and checked by reading, not compiled, not run on hardware
//
#include <hip/hip_runtime.h>
#include <hip/hip_bf16.h>
#include <math.h>

typedef __attribute__((ext_vector_type(16))) _Float16 v16h;
typedef __attribute__((ext_vector_type(8)))  float    v8f;
typedef int gv4i __attribute__((vector_size(16)));   // matches builtin param type

union F16x16 { v16h h; uint4 u[2]; };

#define D_MODEL 1024
#define SEQ_T   2048
#define NB      2
#define NH      16
#define DK      64
#define NROWS   (NB * SEQ_T)   // 4096

// ---------------------------------------------------------------------------
// CDNA5 async global->LDS staging (ASYNCcnt path), with safe fallback
// ---------------------------------------------------------------------------
#if defined(__has_builtin)
#if __has_builtin(__builtin_amdgcn_global_load_async_to_lds_b128)
#define USE_ASYNC_LDS 1
#endif
#endif

__device__ __forceinline__ void stage16(const _Float16* g, _Float16* l) {
#ifdef USE_ASYNC_LDS
    __builtin_amdgcn_global_load_async_to_lds_b128(
        (__attribute__((address_space(1))) gv4i*)g,
        (__attribute__((address_space(3))) gv4i*)l, 0, 0);
#else
    *(uint4*)l = *(const uint4*)g;
#endif
}

__device__ __forceinline__ void stage_wait() {
#ifdef USE_ASYNC_LDS
#if defined(__has_builtin) && __has_builtin(__builtin_amdgcn_s_wait_asynccnt)
    __builtin_amdgcn_s_wait_asynccnt(0);
#else
    asm volatile("s_wait_asynccnt 0x0" ::: "memory");
#endif
#endif
}

// ---------------------------------------------------------------------------
// Kernel 1: quantile MLP embeddings + adaptive-temperature scale
// grid = 6 blocks (b in 0..1, e in 0..2), 256 threads
// ---------------------------------------------------------------------------
__global__ __launch_bounds__(256) void prep_kernel(
    const float* __restrict__ quantile, const float* __restrict__ qimp,
    const float* __restrict__ w1q, const float* __restrict__ b1q,
    const float* __restrict__ w2q, const float* __restrict__ b2q,
    const float* __restrict__ w1k, const float* __restrict__ b1k,
    const float* __restrict__ w2k, const float* __restrict__ b2k,
    const float* __restrict__ w1v, const float* __restrict__ b1v,
    const float* __restrict__ w2v, const float* __restrict__ b2v,
    float* __restrict__ embeds,       // [3][NB][D]
    float* __restrict__ scaleOut)     // [NB]
{
    __shared__ float hbuf[512];
    const int blk = blockIdx.x;
    const int b = blk / 3, e = blk % 3;
    const float* w1 = (e == 0) ? w1q : (e == 1) ? w1k : w1v;
    const float* b1 = (e == 0) ? b1q : (e == 1) ? b1k : b1v;
    const float* w2 = (e == 0) ? w2q : (e == 1) ? w2k : w2v;
    const float* b2 = (e == 0) ? b2q : (e == 1) ? b2k : b2v;
    const float qv = quantile[b];
    const int tid = threadIdx.x;
    for (int j = tid; j < 512; j += 256)
        hbuf[j] = fmaxf(qv * w1[j] + b1[j], 0.0f);
    __syncthreads();
    for (int j = tid; j < D_MODEL; j += 256) {
        float s = b2[j];
        for (int k = 0; k < 512; ++k)
            s += hbuf[k] * w2[(size_t)k * D_MODEL + j];
        embeds[((size_t)e * NB + b) * D_MODEL + j] = s;
    }
    if (blk == 0 && tid == 0) {
        float mx = qimp[0];
        for (int i = 1; i < 100; ++i) mx = fmaxf(mx, qimp[i]);
        for (int bb = 0; bb < NB; ++bb) {
            int idx = (int)(quantile[bb] * 100.0f);
            idx = idx < 0 ? 0 : (idx > 99 ? 99 : idx);
            float imp = qimp[idx];
            if (mx > 0.0f) imp /= mx;
            scaleOut[bb] = 1.0f + imp;   // ADAPTIVE_TEMP = 1
        }
    }
}

// ---------------------------------------------------------------------------
// Kernel 2: weights -> transposed f16  (WT[j][d] = W[d][j])
// grid = (32, 32, 4), block = (32, 8)
// ---------------------------------------------------------------------------
__global__ __launch_bounds__(256) void wtrans_kernel(
    const float* __restrict__ Wq, const float* __restrict__ Wk,
    const float* __restrict__ Wv, const float* __restrict__ Wo,
    _Float16* __restrict__ wt16)
{
    __shared__ float tile[32][33];
    const int z = blockIdx.z;
    const float* W = (z == 0) ? Wq : (z == 1) ? Wk : (z == 2) ? Wv : Wo;
    _Float16* dst = wt16 + (size_t)z * D_MODEL * D_MODEL;
    const int tx = threadIdx.x, ty = threadIdx.y;
    const int x = blockIdx.x * 32 + tx;   // col j of W
    const int y = blockIdx.y * 32 + ty;   // row d of W
#pragma unroll
    for (int k = 0; k < 32; k += 8)
        tile[ty + k][tx] = W[(size_t)(y + k) * D_MODEL + x];
    __syncthreads();
    const int x2 = blockIdx.y * 32 + tx;  // col d of WT
    const int y2 = blockIdx.x * 32 + ty;  // row j of WT
#pragma unroll
    for (int k = 0; k < 32; k += 8)
        dst[(size_t)(y2 + k) * D_MODEL + x2] = (_Float16)tile[tx][ty + k];
}

// ---------------------------------------------------------------------------
// Kernel 3: x + sinusoidal PE  -> f32 (residual) and f16 (GEMM A)
// grid = 4096, block = 256
// ---------------------------------------------------------------------------
__global__ __launch_bounds__(256) void xpe_kernel(
    const float* __restrict__ x, float* __restrict__ xpe32,
    _Float16* __restrict__ xpe16)
{
    const int n = blockIdx.x;
    const int t = n & (SEQ_T - 1);
    for (int j = threadIdx.x; j < D_MODEL; j += 256) {
        int i2 = j & ~1;
        float dv = expf(-9.210340371976184f * (float)i2 / (float)D_MODEL);
        float ang = (float)t * dv;
        float pe = (j & 1) ? cosf(ang) : sinf(ang);
        float v = x[(size_t)n * D_MODEL + j] + pe;
        xpe32[(size_t)n * D_MODEL + j] = v;
        xpe16[(size_t)n * D_MODEL + j] = (_Float16)v;
    }
}

// ---------------------------------------------------------------------------
// Kernel 4: tiled f16 WMMA GEMM, out[n][j] = sum_d A[n][d] * WT[j][d] + bias
// Block tile 128x128, 8 waves (wave w -> rows w*16..+15, all 128 cols), K=32.
// Per wave per K-chunk: 1 A-frag + 8 B-frags (LDS) -> 8 WMMAs.
// mode 0/1: +embed, write Q/K f16 [B,H,T,64]
// mode 2  : +embed, write V  f16 transposed [B,H,64,T]
// mode 3  : +resid (xpe), write f32 y
// ---------------------------------------------------------------------------
__global__ __launch_bounds__(256) void gemm_kernel(
    const _Float16* __restrict__ A,     // [4096][1024] f16
    const _Float16* __restrict__ Wt,    // [1024][1024] f16 (W^T)
    const float* __restrict__ bias,     // [1024]
    const float* __restrict__ embed,    // [NB][1024] or nullptr
    _Float16* __restrict__ outH,
    float* __restrict__ outF,
    const float* __restrict__ resid,
    int mode)
{
    __shared__ __align__(16) _Float16 As[128 * 40];       // 10 KB
    __shared__ __align__(16) _Float16 Bs[128 * 40];       // 10 KB
    __shared__ __align__(16) _Float16 Vbuf[8 * 16 * 128]; // 32 KB (mode 2 only)

    const int tid = threadIdx.x;
    const int w = tid >> 5, lane = tid & 31;
    const int ln16 = lane & 15, hs = lane >> 4;
    const int n0 = blockIdx.x * 128;
    const int j0 = blockIdx.y * 128;
    const int bidx = n0 / SEQ_T;           // whole block in one batch

    v8f acc[8];
#pragma unroll
    for (int ct = 0; ct < 8; ++ct) {
        int j = j0 + ct * 16 + ln16;
        float v = bias[j];
        if (embed) v += embed[(size_t)bidx * D_MODEL + j];
#pragma unroll
        for (int r = 0; r < 8; ++r) acc[ct][r] = v;
    }

    for (int k0 = 0; k0 < D_MODEL; k0 += 32) {
        // stage A tile 128x32 halves and B tile 128x32 halves (16B chunks)
#pragma unroll
        for (int rep = 0; rep < 2; ++rep) {
            int ch = tid + rep * 256;
            int r = ch >> 2, part = ch & 3;
            stage16(A + (size_t)(n0 + r) * D_MODEL + k0 + part * 8,
                    As + r * 40 + part * 8);
            stage16(Wt + (size_t)(j0 + r) * D_MODEL + k0 + part * 8,
                    Bs + r * 40 + part * 8);
        }
        stage_wait();
        __syncthreads();

        F16x16 afr;
        {
            int row = w * 16 + ln16;
            afr.u[0] = *(const uint4*)(As + row * 40 + hs * 8);
            afr.u[1] = *(const uint4*)(As + row * 40 + 16 + hs * 8);
        }
#pragma unroll
        for (int ct = 0; ct < 8; ++ct) {
            F16x16 bfr;
            int jj = ct * 16 + ln16;
            bfr.u[0] = *(const uint4*)(Bs + jj * 40 + hs * 8);
            bfr.u[1] = *(const uint4*)(Bs + jj * 40 + 16 + hs * 8);
            acc[ct] = __builtin_amdgcn_wmma_f32_16x16x32_f16(
                false, afr.h, false, bfr.h, (short)0, acc[ct], false, false);
        }
        __syncthreads();
    }

    const int trowBase = (n0 & (SEQ_T - 1)) + w * 16;
    if (mode <= 1) {
#pragma unroll
        for (int ct = 0; ct < 8; ++ct) {
            const int h = (j0 >> 6) + (ct >> 2);
            const int c = (ct & 3) * 16 + ln16;
#pragma unroll
            for (int r = 0; r < 8; ++r) {
                int t = trowBase + r + hs * 8;
                outH[(((size_t)bidx * NH + h) * SEQ_T + t) * DK + c] =
                    (_Float16)acc[ct][r];
            }
        }
    } else if (mode == 2) {
        _Float16* vb = Vbuf + w * 2048;
#pragma unroll
        for (int ct = 0; ct < 8; ++ct)
#pragma unroll
            for (int r = 0; r < 8; ++r)
                vb[(r + hs * 8) * 128 + ct * 16 + ln16] = (_Float16)acc[ct][r];
        __syncthreads();
#pragma unroll
        for (int rep = 0; rep < 4; ++rep) {
            int cc = lane + rep * 32;              // 0..127 within block cols
            int h = (j0 >> 6) + (cc >> 6);
            int c = cc & 63;
            union { _Float16 hh[16]; uint4 u[2]; } tmp;
#pragma unroll
            for (int i = 0; i < 16; ++i) tmp.hh[i] = vb[i * 128 + cc];
            uint4* dst = (uint4*)(outH +
                (((size_t)bidx * NH + h) * DK + c) * SEQ_T + trowBase);
            dst[0] = tmp.u[0]; dst[1] = tmp.u[1];
        }
    } else {
#pragma unroll
        for (int ct = 0; ct < 8; ++ct)
#pragma unroll
            for (int r = 0; r < 8; ++r) {
                size_t nrow = (size_t)n0 + w * 16 + r + hs * 8;
                int j = j0 + ct * 16 + ln16;
                outF[nrow * D_MODEL + j] = acc[ct][r] + resid[nrow * D_MODEL + j];
            }
    }
}

// ---------------------------------------------------------------------------
// Kernel 5: flash attention, one wave per (b, h, 16-query tile)
// grid = (128, 16, 2), block = 32
// ---------------------------------------------------------------------------
__global__ __launch_bounds__(32) void attn_kernel(
    const _Float16* __restrict__ q16,   // [B,H,T,64]
    const _Float16* __restrict__ k16,   // [B,H,T,64]
    const _Float16* __restrict__ vt16,  // [B,H,64,T]
    const float* __restrict__ scale,    // [B]
    _Float16* __restrict__ o16)         // [B*T][1024]
{
    __shared__ __align__(16) _Float16 pbuf[16 * 40];
    const int lane = threadIdx.x;
    const int ln16 = lane & 15, hs = lane >> 4;
    const int tb = blockIdx.x, h = blockIdx.y, b = blockIdx.z;
    const float sb = scale[b] * 0.125f;   // scale / sqrt(64)

    const _Float16* qrow = q16 + (((size_t)b * NH + h) * SEQ_T + tb * 16 + ln16) * DK;
    F16x16 aQ[2];
#pragma unroll
    for (int kc = 0; kc < 2; ++kc) {
        aQ[kc].u[0] = *(const uint4*)(qrow + kc * 32 + hs * 8);
        aQ[kc].u[1] = *(const uint4*)(qrow + kc * 32 + 16 + hs * 8);
    }

    v8f acc[4];
#pragma unroll
    for (int ct = 0; ct < 4; ++ct)
#pragma unroll
        for (int r = 0; r < 8; ++r) acc[ct][r] = 0.0f;
    float mrow[8], lrow[8];
#pragma unroll
    for (int r = 0; r < 8; ++r) { mrow[r] = -1e30f; lrow[r] = 0.0f; }

    const _Float16* kbase = k16 + ((size_t)b * NH + h) * SEQ_T * DK;
    const _Float16* vbase = vt16 + ((size_t)b * NH + h) * DK * SEQ_T;

    for (int s0 = 0; s0 < SEQ_T; s0 += 32) {
        v8f sc[2];
#pragma unroll
        for (int nt = 0; nt < 2; ++nt) {
            v8f s8;
#pragma unroll
            for (int r = 0; r < 8; ++r) s8[r] = 0.0f;
            const _Float16* krow = kbase + (size_t)(s0 + nt * 16 + ln16) * DK;
#pragma unroll
            for (int kc = 0; kc < 2; ++kc) {
                F16x16 bk;
                bk.u[0] = *(const uint4*)(krow + kc * 32 + hs * 8);
                bk.u[1] = *(const uint4*)(krow + kc * 32 + 16 + hs * 8);
                s8 = __builtin_amdgcn_wmma_f32_16x16x32_f16(
                    false, aQ[kc].h, false, bk.h, (short)0, s8, false, false);
            }
            sc[nt] = s8;
        }
        // online softmax update over these 32 keys
#pragma unroll
        for (int r = 0; r < 8; ++r) {
            float v0 = sc[0][r] * sb, v1 = sc[1][r] * sb;
            float mx = fmaxf(v0, v1);
            mx = fmaxf(mx, __shfl_xor(mx, 1));
            mx = fmaxf(mx, __shfl_xor(mx, 2));
            mx = fmaxf(mx, __shfl_xor(mx, 4));
            mx = fmaxf(mx, __shfl_xor(mx, 8));
            float mn = fmaxf(mrow[r], mx);
            float corr = __expf(mrow[r] - mn);
            mrow[r] = mn;
            float p0 = __expf(v0 - mn), p1 = __expf(v1 - mn);
            float ps = p0 + p1;
            ps += __shfl_xor(ps, 1);
            ps += __shfl_xor(ps, 2);
            ps += __shfl_xor(ps, 4);
            ps += __shfl_xor(ps, 8);
            lrow[r] = lrow[r] * corr + ps;
#pragma unroll
            for (int ct = 0; ct < 4; ++ct) acc[ct][r] *= corr;
            int m = r + hs * 8;
            pbuf[m * 40 + ln16]      = (_Float16)p0;
            pbuf[m * 40 + 16 + ln16] = (_Float16)p1;
        }
        __syncthreads();
        F16x16 aP;
        aP.u[0] = *(const uint4*)(pbuf + ln16 * 40 + hs * 16);
        aP.u[1] = *(const uint4*)(pbuf + ln16 * 40 + hs * 16 + 8);
#pragma unroll
        for (int ct = 0; ct < 4; ++ct) {
            F16x16 bv;
            const _Float16* vrow = vbase + (size_t)(ct * 16 + ln16) * SEQ_T + s0 + hs * 16;
            bv.u[0] = ((const uint4*)vrow)[0];
            bv.u[1] = ((const uint4*)vrow)[1];
            acc[ct] = __builtin_amdgcn_wmma_f32_16x16x32_f16(
                false, aP.h, false, bv.h, (short)0, acc[ct], false, false);
        }
        __syncthreads();
    }

#pragma unroll
    for (int ct = 0; ct < 4; ++ct)
#pragma unroll
        for (int r = 0; r < 8; ++r) {
            int t = tb * 16 + r + hs * 8;
            o16[((size_t)b * SEQ_T + t) * D_MODEL + h * DK + ct * 16 + ln16] =
                (_Float16)(acc[ct][r] / lrow[r]);
        }
}

// ---------------------------------------------------------------------------
// Kernel 6: layernorm per row
// grid = 4096, block = 256
// ---------------------------------------------------------------------------
__global__ __launch_bounds__(256) void ln_kernel(
    const float* __restrict__ y, const float* __restrict__ g,
    const float* __restrict__ beta, float* __restrict__ out)
{
    __shared__ float red[256];
    const int n = blockIdx.x;
    const int tid = threadIdx.x;
    const float* row = y + (size_t)n * D_MODEL;
    float vals[4];
    float s = 0.0f;
#pragma unroll
    for (int i = 0; i < 4; ++i) { vals[i] = row[tid + i * 256]; s += vals[i]; }
    red[tid] = s; __syncthreads();
    for (int off = 128; off > 0; off >>= 1) {
        if (tid < off) red[tid] += red[tid + off];
        __syncthreads();
    }
    float mu = red[0] * (1.0f / D_MODEL);
    __syncthreads();
    float vs = 0.0f;
#pragma unroll
    for (int i = 0; i < 4; ++i) { float d = vals[i] - mu; vs += d * d; }
    red[tid] = vs; __syncthreads();
    for (int off = 128; off > 0; off >>= 1) {
        if (tid < off) red[tid] += red[tid + off];
        __syncthreads();
    }
    float rstd = rsqrtf(red[0] * (1.0f / D_MODEL) + 1e-5f);
#pragma unroll
    for (int i = 0; i < 4; ++i) {
        int j = tid + i * 256;
        out[(size_t)n * D_MODEL + j] = (vals[i] - mu) * rstd * g[j] + beta[j];
    }
}

// ---------------------------------------------------------------------------
// Host launcher
// ---------------------------------------------------------------------------
extern "C" void kernel_launch(void* const* d_in, const int* in_sizes, int n_in,
                              void* d_out, int out_size, void* d_ws, size_t ws_size,
                              hipStream_t stream) {
    const float* x        = (const float*)d_in[0];
    const float* quantile = (const float*)d_in[1];
    const float* qimp     = (const float*)d_in[2];
    const float* Wq = (const float*)d_in[3];  const float* bq = (const float*)d_in[4];
    const float* Wk = (const float*)d_in[5];  const float* bk = (const float*)d_in[6];
    const float* Wv = (const float*)d_in[7];  const float* bv = (const float*)d_in[8];
    const float* Wo = (const float*)d_in[9];  const float* bo = (const float*)d_in[10];
    const float* w1q = (const float*)d_in[11]; const float* b1q = (const float*)d_in[12];
    const float* w2q = (const float*)d_in[13]; const float* b2q = (const float*)d_in[14];
    const float* w1k = (const float*)d_in[15]; const float* b1k = (const float*)d_in[16];
    const float* w2k = (const float*)d_in[17]; const float* b2k = (const float*)d_in[18];
    const float* w1v = (const float*)d_in[19]; const float* b1v = (const float*)d_in[20];
    const float* w2v = (const float*)d_in[21]; const float* b2v = (const float*)d_in[22];
    const float* ln_g = (const float*)d_in[23]; const float* ln_b = (const float*)d_in[24];
    float* out = (float*)d_out;

    char* ws = (char*)d_ws;
    const size_t MB = (size_t)1 << 20;
    _Float16* wt16  = (_Float16*)(ws);               //  8 MB: 4 x W^T f16
    float*    xpe32 = (float*)(ws + 8 * MB);         // 16 MB
    _Float16* xpe16 = (_Float16*)(ws + 24 * MB);     //  8 MB
    _Float16* q16   = (_Float16*)(ws + 32 * MB);     //  8 MB
    _Float16* k16   = (_Float16*)(ws + 40 * MB);     //  8 MB
    _Float16* vt16  = (_Float16*)(ws + 48 * MB);     //  8 MB
    _Float16* o16   = (_Float16*)(ws + 56 * MB);     //  8 MB
    float*    y32   = (float*)(ws + 64 * MB);        // 16 MB
    float*    embeds = (float*)(ws + 80 * MB);       // 3*2*1024 f32
    float*    scalep = embeds + 3 * NB * D_MODEL;    // 2 f32

    prep_kernel<<<6, 256, 0, stream>>>(quantile, qimp,
        w1q, b1q, w2q, b2q, w1k, b1k, w2k, b2k, w1v, b1v, w2v, b2v,
        embeds, scalep);

    wtrans_kernel<<<dim3(32, 32, 4), dim3(32, 8), 0, stream>>>(Wq, Wk, Wv, Wo, wt16);

    xpe_kernel<<<NROWS, 256, 0, stream>>>(x, xpe32, xpe16);

    const size_t WSTEP = (size_t)D_MODEL * D_MODEL;
    dim3 ggrid(NROWS / 128, D_MODEL / 128, 1);
    gemm_kernel<<<ggrid, 256, 0, stream>>>(xpe16, wt16 + 0 * WSTEP, bq,
        embeds + 0 * NB * D_MODEL, q16, nullptr, nullptr, 0);
    gemm_kernel<<<ggrid, 256, 0, stream>>>(xpe16, wt16 + 1 * WSTEP, bk,
        embeds + 1 * NB * D_MODEL, k16, nullptr, nullptr, 1);
    gemm_kernel<<<ggrid, 256, 0, stream>>>(xpe16, wt16 + 2 * WSTEP, bv,
        embeds + 2 * NB * D_MODEL, vt16, nullptr, nullptr, 2);

    attn_kernel<<<dim3(SEQ_T / 16, NH, NB), 32, 0, stream>>>(q16, k16, vt16, scalep, o16);

    gemm_kernel<<<ggrid, 256, 0, stream>>>(o16, wt16 + 3 * WSTEP, bo,
        nullptr, nullptr, y32, xpe32, 3);

    ln_kernel<<<NROWS, 256, 0, stream>>>(y32, ln_g, ln_b, out);
}
